// MoChA_46033459478994
// MI455X (gfx1250) — matrix operand order repaired and checked by reading, not compiled, not run on hardware
//
#include <hip/hip_runtime.h>
#include <math.h>

#define B_      8
#define KLEN    3000
#define QLEN    256
#define DMODEL  512
#define HMA     4
#define DHEAD   128
#define EPS_    1e-6f
#define INV_SQRT_D 0.04419417382415922f   // 1/sqrt(512)

typedef __attribute__((ext_vector_type(16))) __bf16 v16bf;
typedef __attribute__((ext_vector_type(8)))  float  v8f;
typedef __attribute__((ext_vector_type(4)))  unsigned int u32x4;
typedef __attribute__((ext_vector_type(8)))  int  i32x8;
typedef __attribute__((ext_vector_type(4)))  int  i32x4;

__device__ __forceinline__ __bf16 f2bf(float x) { return (__bf16)x; }

// ---------------------------------------------------------------------------
// TDM helper: stage a 32(k) x 64(n) f32 tile of W (row stride DMODEL) into
// LDS at byte offset lds_off, via the Tensor Data Mover (TENSORcnt tracked).
// Descriptor per CDNA5 ISA 8.3/8.4 (2D tensor: groups 2/3 zero).
// ---------------------------------------------------------------------------
#define PROJ_ROWS 128          // 8 waves x 16
#define PROJ_COLS 64
#define PROJ_KK   32
#define PROJ_TILE_BYTES (PROJ_KK * PROJ_COLS * 4)   // 8192

__device__ __forceinline__ void tdm_load_w_tile(const float* gsrc,
                                                unsigned int lds_off)
{
    unsigned long long ga = (unsigned long long)(uintptr_t)gsrc;
    u32x4 g0;
    g0[0] = 1u;                                          // count=1, user mode
    g0[1] = lds_off;                                     // lds_addr (bytes)
    g0[2] = (unsigned int)(ga & 0xffffffffull);          // global_addr[31:0]
    g0[3] = (unsigned int)((ga >> 32) & 0x1ffffffull)    // global_addr[56:32]
            | (2u << 30);                                // type = 2 ("image")
    i32x8 g1;
    g1[0] = (int)(2u << 16);                             // data_size = 4 bytes
    g1[1] = (int)(((unsigned)DMODEL & 0xffffu) << 16);   // tensor_dim0 lo16 (=512)
    g1[2] = (int)(((unsigned)DMODEL >> 16)               // tensor_dim0 hi16
            | (((unsigned)DMODEL & 0xffffu) << 16));     // tensor_dim1 lo16 (=512)
    g1[3] = (int)(((unsigned)DMODEL >> 16)               // tensor_dim1 hi16
            | ((unsigned)PROJ_COLS << 16));              // tile_dim0 = 64
    g1[4] = (int)(unsigned)PROJ_KK;                      // tile_dim1 = 32, tile_dim2 = 0
    g1[5] = (int)(unsigned)DMODEL;                       // tensor_dim0_stride = 512
    g1[6] = 0;                                           // stride0 hi / stride1 lo
    g1[7] = 0;                                           // stride1 hi
    i32x4 z4 = {};
#if defined(__clang_major__) && (__clang_major__ >= 23)
    i32x8 z8 = {};
    __builtin_amdgcn_tensor_load_to_lds(g0, g1, z4, z4, z8, 0);
#else
    __builtin_amdgcn_tensor_load_to_lds(g0, g1, z4, z4, 0);
#endif
}

// ---------------------------------------------------------------------------
// Projection GEMM: out[M x 512](bf16) = A[M x 512](f32) @ W[512 x 512](f32) + b
// Block = 8 waves = 128x64 C tile. W k-panel staged in LDS by TDM, double
// buffered so the DMA overlaps WMMA; each wave: 1 A fragment -> 4 WMMAs.
// ---------------------------------------------------------------------------
__global__ void __launch_bounds__(256)
proj_gemm_tdm(const float* __restrict__ A,
              const float* __restrict__ W,
              const float* __restrict__ bias,
              __bf16* __restrict__ out, int M)
{
    extern __shared__ float smemW[];                      // 2 x 32 x 64 f32

    const int tid   = threadIdx.x;
    const int lane  = tid & 31;
    const int wid   = tid >> 5;
    const int half  = lane >> 4;
    const int lan15 = lane & 15;

    const int nBlkN = DMODEL / PROJ_COLS;                 // 8
    const int brow0 = (blockIdx.x / nBlkN) * PROJ_ROWS;
    const int bcol0 = (blockIdx.x % nBlkN) * PROJ_COLS;

    const int arow   = brow0 + wid * 16 + lan15;          // A row for this lane
    const int arow_c = (arow < M) ? arow : (M - 1);       // clamp (stores guarded)

    v8f acc[4] = {};

    const int NIT = DMODEL / PROJ_KK;                     // 16
    if (wid == 0)
        tdm_load_w_tile(W + 0 * (size_t)DMODEL + bcol0, 0u);

    for (int it = 0; it < NIT; ++it) {
        const int k0 = it * PROJ_KK;
        if (wid == 0)
            __builtin_amdgcn_s_wait_tensorcnt(0);         // tile[it&1] landed
        __syncthreads();
        if (wid == 0 && (it + 1) < NIT)                   // prefetch next panel
            tdm_load_w_tile(W + (size_t)(k0 + PROJ_KK) * DMODEL + bcol0,
                            (unsigned)(((it + 1) & 1) * PROJ_TILE_BYTES));

        // A fragment: 16x32 bf16 (lane row = lan15, K pairs per VGPR)
        v16bf a;
        #pragma unroll
        for (int v = 0; v < 8; ++v) {
            int ka = 2 * v + half * 16;
            const float* ap = A + (size_t)arow_c * DMODEL + k0 + ka;
            a[2 * v + 0] = f2bf(ap[0]);
            a[2 * v + 1] = f2bf(ap[1]);
        }
        const float* Wt = smemW + (it & 1) * (PROJ_KK * PROJ_COLS);
        #pragma unroll
        for (int ct = 0; ct < 4; ++ct) {
            v16bf b;
            #pragma unroll
            for (int v = 0; v < 8; ++v) {
                int kb = 2 * v + half * 16;               // B rows {2v,2v+1} (+16 hi)
                b[2 * v + 0] = f2bf(Wt[(kb + 0) * PROJ_COLS + ct * 16 + lan15]);
                b[2 * v + 1] = f2bf(Wt[(kb + 1) * PROJ_COLS + ct * 16 + lan15]);
            }
            acc[ct] = __builtin_amdgcn_wmma_f32_16x16x32_bf16(
                false, a, false, b, (short)0, acc[ct], false, false);
        }
        __syncthreads();                                  // reads done before overwrite
    }

    #pragma unroll
    for (int ct = 0; ct < 4; ++ct) {
        float bv = bias[bcol0 + ct * 16 + lan15];
        #pragma unroll
        for (int v = 0; v < 8; ++v) {
            int m = brow0 + wid * 16 + v + half * 8;      // C row layout
            if (m < M)
                out[(size_t)m * DMODEL + bcol0 + ct * 16 + lan15] =
                    f2bf(acc[ct][v] + bv);
        }
    }
}

// ---------------------------------------------------------------------------
// p_choose[b,h,q,k] = sigmoid(q_ma . k_ma / sqrt(512) + r[h])   (d = 128)
// ---------------------------------------------------------------------------
__global__ void pchoose_wmma(const __bf16* __restrict__ qma,
                             const __bf16* __restrict__ kma,
                             const float*  __restrict__ r,
                             float* __restrict__ pout)
{
    const int lane = threadIdx.x & 31;
    const int wid  = threadIdx.x >> 5;
    const int KT   = (KLEN + 15) / 16;                // 188
    int tile = blockIdx.x * (blockDim.x >> 5) + wid;
    if (tile >= B_ * HMA * (QLEN / 16) * KT) return;
    int kt = tile % KT;            tile /= KT;
    int qt = tile % (QLEN / 16);   tile /= (QLEN / 16);
    int h  = tile % HMA;
    int b  = tile / HMA;

    const int half = lane >> 4, lan15 = lane & 15;
    const int kcol = kt * 16 + lan15;
    const bool okc = (kcol < KLEN);

    v8f acc = {};
    for (int k0 = 0; k0 < DHEAD; k0 += 32) {
        v16bf a, bm;
        #pragma unroll
        for (int v = 0; v < 8; ++v) {
            int ka = 2 * v + half * 16;
            const __bf16* ap = qma + (size_t)(b * QLEN + qt * 16 + lan15) * DMODEL
                                   + h * DHEAD + k0 + ka;
            a[2 * v + 0] = ap[0];
            a[2 * v + 1] = ap[1];
            int kb = 2 * v + half * 16;
            const __bf16* bp = kma + (size_t)(b * KLEN + (okc ? kcol : 0)) * DMODEL
                                   + h * DHEAD + k0 + kb;
            __bf16 z = f2bf(0.0f);
            bm[2 * v + 0] = okc ? bp[0] : z;
            bm[2 * v + 1] = okc ? bp[1] : z;
        }
        acc = __builtin_amdgcn_wmma_f32_16x16x32_bf16(false, a, false, bm,
                                                      (short)0, acc, false, false);
    }
    float rh = r[h];
    #pragma unroll
    for (int v = 0; v < 8; ++v) {
        int q = qt * 16 + v + half * 8;
        if (okc) {
            float e = acc[v] * INV_SQRT_D + rh;
            pout[((size_t)(b * HMA + h) * QLEN + q) * KLEN + kcol] =
                1.0f / (1.0f + __expf(-e));
        }
    }
}

// ---------------------------------------------------------------------------
// u[b,q,k] = q_ca . k_ca / sqrt(512)   (single CA head, d = 512)
// ---------------------------------------------------------------------------
__global__ void u_wmma(const __bf16* __restrict__ qca,
                       const __bf16* __restrict__ kca,
                       float* __restrict__ uout)
{
    const int lane = threadIdx.x & 31;
    const int wid  = threadIdx.x >> 5;
    const int KT   = (KLEN + 15) / 16;
    int tile = blockIdx.x * (blockDim.x >> 5) + wid;
    if (tile >= B_ * (QLEN / 16) * KT) return;
    int kt = tile % KT;          tile /= KT;
    int qt = tile % (QLEN / 16); tile /= (QLEN / 16);
    int b  = tile;

    const int half = lane >> 4, lan15 = lane & 15;
    const int kcol = kt * 16 + lan15;
    const bool okc = (kcol < KLEN);

    v8f acc = {};
    for (int k0 = 0; k0 < DMODEL; k0 += 32) {
        v16bf a, bm;
        #pragma unroll
        for (int v = 0; v < 8; ++v) {
            int ka = 2 * v + half * 16;
            const __bf16* ap = qca + (size_t)(b * QLEN + qt * 16 + lan15) * DMODEL + k0 + ka;
            a[2 * v + 0] = ap[0];
            a[2 * v + 1] = ap[1];
            int kb = 2 * v + half * 16;
            const __bf16* bp = kca + (size_t)(b * KLEN + (okc ? kcol : 0)) * DMODEL + k0 + kb;
            __bf16 z = f2bf(0.0f);
            bm[2 * v + 0] = okc ? bp[0] : z;
            bm[2 * v + 1] = okc ? bp[1] : z;
        }
        acc = __builtin_amdgcn_wmma_f32_16x16x32_bf16(false, a, false, bm,
                                                      (short)0, acc, false, false);
    }
    #pragma unroll
    for (int v = 0; v < 8; ++v) {
        int q = qt * 16 + v + half * 8;
        if (okc)
            uout[((size_t)b * QLEN + q) * KLEN + kcol] = acc[v] * INV_SQRT_D;
    }
}

// ---------------------------------------------------------------------------
// Monotonic alpha recurrence: one block per (b,h); sequential over Q=256,
// two hierarchical block scans over K=3000 per step (3 elems/thread + LDS).
// ---------------------------------------------------------------------------
#define SCAN_NT 1024

__device__ __forceinline__ float block_exscan(float v, float* sc, int tid)
{
    sc[tid] = v;
    __syncthreads();
    #pragma unroll 1
    for (int s = 1; s < SCAN_NT; s <<= 1) {
        float t = (tid >= s) ? sc[tid - s] : 0.0f;
        __syncthreads();
        sc[tid] += t;
        __syncthreads();
    }
    float incl = sc[tid];
    __syncthreads();
    return incl - v;                                   // exclusive offset
}

__global__ void __launch_bounds__(SCAN_NT)
alpha_scan(const float* __restrict__ p_all, float* __restrict__ alpha_all)
{
    const int bh   = blockIdx.x;                       // b*HMA + h
    const int tid  = threadIdx.x;
    const int base = tid * 3;
    __shared__ float sc[SCAN_NT];

    const float* prow_base = p_all     + (size_t)bh * QLEN * KLEN;
    float*       arow_base = alpha_all + (size_t)bh * QLEN * KLEN;

    float ap0 = (base == 0) ? 1.0f : 0.0f, ap1 = 0.0f, ap2 = 0.0f;   // a0 one-hot

    for (int q = 0; q < QLEN; ++q) {
        const float* prow = prow_base + (size_t)q * KLEN;
        float p0 = (base + 0 < KLEN) ? prow[base + 0] : 0.0f;
        float p1 = (base + 1 < KLEN) ? prow[base + 1] : 0.0f;
        float p2 = (base + 2 < KLEN) ? prow[base + 2] : 0.0f;

        // cp = exp(exclusive_cumsum(log(clip(1-p, eps, 1))))
        float x0 = __logf(fminf(fmaxf(1.0f - p0, EPS_), 1.0f));
        float x1 = __logf(fminf(fmaxf(1.0f - p1, EPS_), 1.0f));
        float x2 = __logf(fminf(fmaxf(1.0f - p2, EPS_), 1.0f));
        float s0 = x0, s1 = x0 + x1, s2 = s1 + x2;
        float off = block_exscan(s2, sc, tid);
        float cp0 = __expf(off + s0 - x0);
        float cp1 = __expf(off + s1 - x1);
        float cp2 = __expf(off + s2 - x2);
        float c0 = fminf(fmaxf(cp0, EPS_), 1.0f);
        float c1 = fminf(fmaxf(cp1, EPS_), 1.0f);
        float c2 = fminf(fmaxf(cp2, EPS_), 1.0f);

        // a_q = p * cp * inclusive_cumsum(a_prev / cp_clipped)
        float t0 = ap0 / c0, t1 = ap1 / c1, t2 = ap2 / c2;
        float u0 = t0, u1 = t0 + t1, u2 = u1 + t2;
        float off2 = block_exscan(u2, sc, tid);
        float a0 = p0 * cp0 * (off2 + u0);
        float a1 = p1 * cp1 * (off2 + u1);
        float a2 = p2 * cp2 * (off2 + u2);

        float* arow = arow_base + (size_t)q * KLEN;
        if (base + 0 < KLEN) arow[base + 0] = a0;
        if (base + 1 < KLEN) arow[base + 1] = a1;
        if (base + 2 < KLEN) arow[base + 2] = a2;
        ap0 = a0; ap1 = a1; ap2 = a2;
    }
}

// ---------------------------------------------------------------------------
// Chunkwise beta: one block per (b,q); softmax-exp, 16-tap moving sums in LDS.
// ---------------------------------------------------------------------------
__global__ void __launch_bounds__(512)
beta_kernel(const float* __restrict__ u_all,
            const float* __restrict__ alpha_all,
            float* __restrict__ beta_all)
{
    const int bq  = blockIdx.x;
    const int b   = bq / QLEN, q = bq % QLEN;
    const int tid = threadIdx.x;
    const int NT  = blockDim.x;                        // 512
    __shared__ float ue[KLEN];
    __shared__ float dn[KLEN];
    __shared__ float w[KLEN];
    __shared__ float red[512];

    const float* urow = u_all + ((size_t)b * QLEN + q) * KLEN;
    float mx = -1e30f;
    for (int k = tid; k < KLEN; k += NT) mx = fmaxf(mx, urow[k]);
    red[tid] = mx;
    __syncthreads();
    for (int s = NT >> 1; s > 0; s >>= 1) {
        if (tid < s) red[tid] = fmaxf(red[tid], red[tid + s]);
        __syncthreads();
    }
    mx = red[0];
    for (int k = tid; k < KLEN; k += NT)
        ue[k] = fmaxf(__expf(urow[k] - mx), 1e-5f);    // clip(exp, 1e-5, inf)
    __syncthreads();
    for (int k = tid; k < KLEN; k += NT) {             // denom: window [k-15, k]
        float s = 0.0f;
        int j0 = k - 15; if (j0 < 0) j0 = 0;
        for (int j = j0; j <= k; ++j) s += ue[j];
        dn[k] = s;
    }
    __syncthreads();
    for (int h = 0; h < HMA; ++h) {
        const float* arow = alpha_all + ((size_t)(b * HMA + h) * QLEN + q) * KLEN;
        for (int k = tid; k < KLEN; k += NT) w[k] = arow[k] / dn[k];
        __syncthreads();
        float* brow = beta_all + ((size_t)(b * HMA + h) * QLEN + q) * KLEN;
        for (int k = tid; k < KLEN; k += NT) {         // window [k, k+15]
            float s = 0.0f;
            int j1 = k + 15; if (j1 >= KLEN) j1 = KLEN - 1;
            for (int j = k; j <= j1; ++j) s += w[j];
            brow[k] = ue[k] * s;
        }
        __syncthreads();
    }
}

// ---------------------------------------------------------------------------
// cv[b,q,h*128+d] = sum_k beta[b,h,q,k] * v[b,k,h*128+d]   (K = 3000, padded)
// ---------------------------------------------------------------------------
__global__ void cv_wmma(const float* __restrict__ beta_all,
                        const __bf16* __restrict__ vproj,
                        float* __restrict__ cv)
{
    const int lane = threadIdx.x & 31;
    const int wid  = threadIdx.x >> 5;
    int tile = blockIdx.x * (blockDim.x >> 5) + wid;
    const int NTN = DHEAD / 16;                        // 8
    const int NTQ = QLEN / 16;                         // 16
    int nt = tile % NTN; tile /= NTN;
    int qt = tile % NTQ; tile /= NTQ;
    int h  = tile % HMA;
    int b  = tile / HMA;

    const int half = lane >> 4, lan15 = lane & 15;
    const float* bmat = beta_all + (size_t)(b * HMA + h) * QLEN * KLEN;

    v8f acc = {};
    for (int k0 = 0; k0 < KLEN; k0 += 32) {
        v16bf a, bm;
        #pragma unroll
        for (int v = 0; v < 8; ++v) {
            int ka   = 2 * v + half * 16;
            int krow = k0 + ka;
            const float* ap = bmat + (size_t)(qt * 16 + lan15) * KLEN + krow;
            a[2 * v + 0] = f2bf((krow     < KLEN) ? ap[0] : 0.0f);
            a[2 * v + 1] = f2bf((krow + 1 < KLEN) ? ap[1] : 0.0f);
            int kb = k0 + 2 * v + half * 16;
            const __bf16* bp = vproj + (size_t)(b * KLEN + ((kb < KLEN) ? kb : 0)) * DMODEL
                                     + h * DHEAD + nt * 16 + lan15;
            __bf16 z = f2bf(0.0f);
            bm[2 * v + 0] = (kb     < KLEN) ? bp[0]      : z;
            bm[2 * v + 1] = (kb + 1 < KLEN) ? bp[DMODEL] : z;
        }
        acc = __builtin_amdgcn_wmma_f32_16x16x32_bf16(false, a, false, bm,
                                                      (short)0, acc, false, false);
    }
    #pragma unroll
    for (int v = 0; v < 8; ++v) {
        int q = qt * 16 + v + half * 8;
        cv[((size_t)(b * QLEN) + q) * DMODEL + h * DHEAD + nt * 16 + lan15] = acc[v];
    }
}

// ---------------------------------------------------------------------------
extern "C" void kernel_launch(void* const* d_in, const int* in_sizes, int n_in,
                              void* d_out, int out_size, void* d_ws, size_t ws_size,
                              hipStream_t stream)
{
    const float* key        = (const float*)d_in[0];
    const float* query      = (const float*)d_in[1];
    const float* w_key_ma   = (const float*)d_in[2];
    const float* b_key_ma   = (const float*)d_in[3];
    const float* w_query_ma = (const float*)d_in[4];
    const float* b_query_ma = (const float*)d_in[5];
    const float* r          = (const float*)d_in[6];
    const float* w_key_ca   = (const float*)d_in[7];
    const float* b_key_ca   = (const float*)d_in[8];
    const float* w_query_ca = (const float*)d_in[9];
    const float* b_query_ca = (const float*)d_in[10];
    const float* w_value    = (const float*)d_in[11];
    const float* b_value    = (const float*)d_in[12];

    float* out   = (float*)d_out;
    float* cv    = out;                                          // [B,Q,512]
    float* alpha = cv + (size_t)B_ * QLEN * DMODEL;              // [B,4,Q,K]
    float* beta  = alpha + (size_t)B_ * HMA * QLEN * KLEN;       // [B,4,Q,K]

    char* ws = (char*)d_ws;
    size_t off = 0;
    __bf16* kma = (__bf16*)(ws + off); off += (size_t)B_ * KLEN * DMODEL * 2;
    __bf16* kca = (__bf16*)(ws + off); off += (size_t)B_ * KLEN * DMODEL * 2;
    __bf16* vbf = (__bf16*)(ws + off); off += (size_t)B_ * KLEN * DMODEL * 2;
    __bf16* qma = (__bf16*)(ws + off); off += (size_t)B_ * QLEN * DMODEL * 2;
    __bf16* qca = (__bf16*)(ws + off); off += (size_t)B_ * QLEN * DMODEL * 2;
    float*  u   = (float*)(ws + off);  off += (size_t)B_ * QLEN * KLEN * 4;

    // 1) projections (bf16 WMMA GEMMs, W panels staged by TDM, double-buffered)
    {
        const size_t shmem = 2 * PROJ_TILE_BYTES;                // 16 KB
        int nBlkN = DMODEL / PROJ_COLS;                          // 8
        int Mk = B_ * KLEN;                                      // 24000
        int gK = ((Mk + PROJ_ROWS - 1) / PROJ_ROWS) * nBlkN;     // 188*8 = 1504
        proj_gemm_tdm<<<gK, 256, shmem, stream>>>(key, w_key_ma, b_key_ma, kma, Mk);
        proj_gemm_tdm<<<gK, 256, shmem, stream>>>(key, w_key_ca, b_key_ca, kca, Mk);
        proj_gemm_tdm<<<gK, 256, shmem, stream>>>(key, w_value,  b_value,  vbf, Mk);
        int Mq = B_ * QLEN;                                      // 2048
        int gQ = ((Mq + PROJ_ROWS - 1) / PROJ_ROWS) * nBlkN;     // 128
        proj_gemm_tdm<<<gQ, 256, shmem, stream>>>(query, w_query_ma, b_query_ma, qma, Mq);
        proj_gemm_tdm<<<gQ, 256, shmem, stream>>>(query, w_query_ca, b_query_ca, qca, Mq);
    }
    // 2) p_choose -> staged in the (not-yet-written) beta output region
    {
        int KT = (KLEN + 15) / 16;
        int tiles = B_ * HMA * (QLEN / 16) * KT;                 // 96256
        pchoose_wmma<<<(tiles + 7) / 8, 256, 0, stream>>>(qma, kma, r, beta);
    }
    // 3) chunkwise energies u
    {
        int KT = (KLEN + 15) / 16;
        int tiles = B_ * (QLEN / 16) * KT;                       // 24064
        u_wmma<<<(tiles + 7) / 8, 256, 0, stream>>>(qca, kca, u);
    }
    // 4) monotonic alpha (sequential in Q)
    alpha_scan<<<B_ * HMA, SCAN_NT, 0, stream>>>(beta, alpha);
    // 5) beta (overwrites the staged p_choose)
    beta_kernel<<<B_ * QLEN, 512, 0, stream>>>(u, alpha, beta);
    // 6) cv = beta @ v
    {
        int tiles = B_ * HMA * (QLEN / 16) * (DHEAD / 16);       // 4096
        cv_wmma<<<tiles / 8, 256, 0, stream>>>(beta, vbf, cv);
    }
}
